// RelPartialLearnableMultiHeadAttn_FishPP_35699768165042
// MI455X (gfx1250) — compile-verified, matrix operand order checked by reading, not run on hardware
//
#include <hip/hip_runtime.h>

// ---------------------------------------------------------------------------
// Types / helpers
// ---------------------------------------------------------------------------
typedef float  v8f   __attribute__((ext_vector_type(8)));
typedef __bf16 v16bf __attribute__((ext_vector_type(16)));

union BF16Frag { unsigned u[8]; v16bf v; };
union F32x8    { float f[8];   v8f   v; };

// Pack two floats to packed bf16 (low = a, high = b).
#if __has_builtin(__builtin_amdgcn_cvt_pk_bf16_f32)
__device__ __forceinline__ unsigned pack2bf(float a, float b) {
  auto p = __builtin_amdgcn_cvt_pk_bf16_f32(a, b);
  unsigned u = 0;
  __builtin_memcpy(&u, &p, sizeof(p) < 4 ? sizeof(p) : 4);
  return u;
}
#else
// Toolchain lacks the builtin; emit the CDNA5 instruction directly.
__device__ __forceinline__ unsigned pack2bf(float a, float b) {
  unsigned out;
  asm("v_cvt_pk_bf16_f32 %0, %1, %2" : "=v"(out) : "v"(a), "v"(b));
  return out;
}
#endif

__device__ __forceinline__ unsigned short f2bf(float f) {
  return (unsigned short)(pack2bf(f, 0.f) & 0xFFFFu);
}
__device__ __forceinline__ float bf2f(unsigned short h) {
  union { unsigned u; float f; } x; x.u = ((unsigned)h) << 16;
  return x.f;
}

// WMMA fragment K-index for 16-bit A(16x32)/B(32x16), per CDNA5 ISA layout:
// half-pair hp (0..7), lane-half hi -> K = ((hp>=4)?16:0) + hi*8 + (hp&3)*2
__device__ __forceinline__ int frag_k(int hp, int hi) {
  return ((hp >= 4) ? 16 : 0) + (hi << 3) + ((hp & 3) << 1);
}

// ---------------------------------------------------------------------------
// Generic batched GEMM: C[z] = A[z] (MxK, fp32) * B[(z/bDiv)%bMod]^T (NxK, fp32)
// bf16 WMMA 16x16x32, f32 accumulate. Block tile 128x128, 8 waves (4Mx2N),
// wave tile 32x64 (2x4 subtiles). mode==1: add rel-shifted BD gather.
// ---------------------------------------------------------------------------
__global__ __launch_bounds__(256)
void fishpp_gemm_bf16(const float* __restrict__ A, long long sA,
                      const float* __restrict__ B, long long sB, int bDiv, int bMod,
                      float* __restrict__ C, long long sC,
                      int M, int N, int K,
                      int mode, const float* __restrict__ bdraw, long long sBD)
{
  const int z = blockIdx.z;
  A += (size_t)z * (size_t)sA;
  B += (size_t)((z / bDiv) % bMod) * (size_t)sB;
  C += (size_t)z * (size_t)sC;

  __shared__ alignas(16) unsigned short At[128][36]; // rows of A tile, +pad
  __shared__ alignas(16) unsigned short Bt[128][36]; // rows of B (N-major) tile

  const int tid  = threadIdx.x;
  const int wave = tid >> 5, lane = tid & 31;
  const int wm = wave & 3, wn = wave >> 2;           // waveM 0..3, waveN 0..1
  const int m0 = blockIdx.y * 128, n0 = blockIdx.x * 128;
  const int l15 = lane & 15, hi = lane >> 4;

  F32x8 acc[2][4];
  #pragma unroll
  for (int mi = 0; mi < 2; ++mi)
    #pragma unroll
    for (int ni = 0; ni < 4; ++ni)
      #pragma unroll
      for (int e = 0; e < 8; ++e) acc[mi][ni].f[e] = 0.f;

  const int lrow = tid >> 3;           // 0..31
  const int lcol = (tid & 7) * 4;      // 0..28

  for (int k0 = 0; k0 < K; k0 += 32) {
    // Phase 1: issue all tile loads back-to-back (let loadcnt waits stage).
    float4 a4[4], b4[4];
    #pragma unroll
    for (int it = 0; it < 4; ++it) {
      const int r = lrow + 32 * it;
      a4[it] = *(const float4*)(A + (size_t)(m0 + r) * K + k0 + lcol);
      b4[it] = *(const float4*)(B + (size_t)(n0 + r) * K + k0 + lcol);
    }
    if (k0 + 32 < K) {                 // prefetch next k-tile (global_prefetch_b8)
      __builtin_prefetch(A + (size_t)(m0 + lrow) * K + k0 + 32 + lcol, 0, 1);
      __builtin_prefetch(B + (size_t)(n0 + lrow) * K + k0 + 32 + lcol, 0, 1);
    }
    __syncthreads();                   // previous iter's frag reads are done
    // Phase 2: convert (v_cvt_pk_bf16_f32) and store to LDS.
    #pragma unroll
    for (int it = 0; it < 4; ++it) {
      const int r = lrow + 32 * it;
      uint2 pa, pb;
      pa.x = pack2bf(a4[it].x, a4[it].y);
      pa.y = pack2bf(a4[it].z, a4[it].w);
      pb.x = pack2bf(b4[it].x, b4[it].y);
      pb.y = pack2bf(b4[it].z, b4[it].w);
      *(uint2*)&At[r][lcol] = pa;
      *(uint2*)&Bt[r][lcol] = pb;
    }
    __syncthreads();

    BF16Frag afr[2], bfr[4];
    #pragma unroll
    for (int mi = 0; mi < 2; ++mi) {
      const int rr = wm * 32 + mi * 16 + l15;
      #pragma unroll
      for (int hp = 0; hp < 8; ++hp)
        afr[mi].u[hp] = *(const unsigned*)&At[rr][frag_k(hp, hi)];
    }
    #pragma unroll
    for (int ni = 0; ni < 4; ++ni) {
      const int rr = wn * 64 + ni * 16 + l15;
      #pragma unroll
      for (int hp = 0; hp < 8; ++hp)
        bfr[ni].u[hp] = *(const unsigned*)&Bt[rr][frag_k(hp, hi)];
    }
    #pragma unroll
    for (int mi = 0; mi < 2; ++mi)
      #pragma unroll
      for (int ni = 0; ni < 4; ++ni)
        acc[mi][ni].v = __builtin_amdgcn_wmma_f32_16x16x32_bf16(
            false, afr[mi].v, false, bfr[ni].v, (short)0, acc[mi][ni].v, false, false);
  }

  // Epilogue: C/D layout -> element (M = r8 + 8*hi, N = lane&15)
  const float* bdz = bdraw + (size_t)z * (size_t)sBD;
  #pragma unroll
  for (int mi = 0; mi < 2; ++mi)
    #pragma unroll
    for (int ni = 0; ni < 4; ++ni)
      #pragma unroll
      for (int r8 = 0; r8 < 8; ++r8) {
        const int gm = m0 + wm * 32 + mi * 16 + r8 + 8 * hi;
        const int gn = n0 + wn * 64 + ni * 16 + l15;
        float v = acc[mi][ni].f[r8];
        if (mode == 1) {
          // Transformer-XL rel_shift gather from raw BD:
          //  j<=i  : raw[i, N-1-i+j] ; j==i+1 : 0 ; j>i+1 : raw[i+1, j-i-2]
          float add = 0.f;
          if (gn <= gm)          add = bdz[(size_t)gm * N + (N - 1 - gm + gn)];
          else if (gn > gm + 1)  add = bdz[(size_t)(gm + 1) * N + (gn - gm - 2)];
          v += add;
        }
        C[(size_t)gm * N + gn] = v;
      }
}

// ---------------------------------------------------------------------------
// Scatter w_heads -> q(+r_w_bias), q(+r_r_bias), k (fp32 [b,g,i,64]) and
// V transposed bf16 [b,h,64,1024]
// ---------------------------------------------------------------------------
__global__ __launch_bounds__(256)
void fishpp_scatter_qkv(const float* __restrict__ C1,
                        const float* __restrict__ rwb, const float* __restrict__ rrb,
                        float* __restrict__ qrw, float* __restrict__ qrr,
                        float* __restrict__ kbuf, unsigned short* __restrict__ vT)
{
  const size_t tid = (size_t)blockIdx.x * 256 + threadIdx.x; // 4096*1280
  const int col = (int)(tid % 1280);
  const size_t row = tid / 1280;
  const int i = (int)(row >> 2), b = (int)(row & 3);
  const float v = C1[tid];
  if (col < 128) {
    const int g = col >> 6, d = col & 63;
    const size_t o = ((size_t)(b * 2 + g) * 1024 + i) * 64 + d;
    qrw[o] = v + rwb[g * 64 + d];
    qrr[o] = v + rrb[g * 64 + d];
  } else if (col < 256) {
    const int c = col - 128, g = c >> 6, d = c & 63;
    kbuf[((size_t)(b * 2 + g) * 1024 + i) * 64 + d] = v;
  } else {
    const int c = col - 256, h = c >> 6, d = c & 63;
    vT[((size_t)(b * 16 + h) * 64 + d) * 1024 + i] = f2bf(v);
  }
}

__global__ __launch_bounds__(256)
void fishpp_scatter_r(const float* __restrict__ C2, float* __restrict__ rk)
{
  const int tid = blockIdx.x * 256 + threadIdx.x;  // 1024*128
  const int col = tid & 127, j = tid >> 7;
  const int g = col >> 6, d = col & 63;
  rk[((size_t)g * 1024 + j) * 64 + d] = C2[tid];
}

// mw[h][i][j] = SCALE * sum_l masks[i,j,l] * mask_proj[l,h]   (bf16)
__global__ __launch_bounds__(256)
void fishpp_mw(const float* __restrict__ masks, const float* __restrict__ mproj,
               unsigned short* __restrict__ mw)
{
  const size_t tid = (size_t)blockIdx.x * 256 + threadIdx.x; // 16M
  const int h = (int)(tid >> 20);
  const size_t rem = tid & 0xFFFFFu;
  const float* mk = masks + rem * 3;
  const float v = mk[0] * mproj[0 * 16 + h] + mk[1] * mproj[1 * 16 + h] + mk[2] * mproj[2 * 16 + h];
  mw[tid] = f2bf(v * 0.125f);                      // SCALE = 1/sqrt(64)
}

// ---------------------------------------------------------------------------
// Fused mask-mix + softmax + PV. Block = (b, g, 16-row i-block); 8 waves = 8
// heads sharing the S block. outT = V^T x P^T via bf16 WMMA. V tiles staged
// to LDS with CDNA5 async global->LDS copies (ASYNCcnt).
// ---------------------------------------------------------------------------
__global__ __launch_bounds__(256)
void fishpp_attn(const float* __restrict__ S,            // [8][1024][1024]
                 const unsigned short* __restrict__ mw,  // [16][1024][1024] bf16
                 const unsigned short* __restrict__ vT,  // [4][16][64][1024] bf16
                 float* __restrict__ attn_vec)           // [4096][1024] fp32
{
  const int ib = blockIdx.x, g = blockIdx.y, b = blockIdx.z;
  const int wave = threadIdx.x >> 5, lane = threadIdx.x & 31;
  const int head = g * 8 + wave;
  const int i0 = ib * 16;
  const int r = lane & 15, c0 = lane >> 4;

  const float*          sr = S  + ((size_t)(b * 2 + g) * 1024 + i0 + r) * 1024;
  const unsigned short* mr = mw + ((size_t)head * 1024 + i0 + r) * 1024;

  __shared__ alignas(16) unsigned short Pf[8][16][36];
  __shared__ alignas(16) unsigned short Vt[8][64][40];
  __shared__ float rowsum[8][16];

  // Pass 1: online row max / exp-sum (lane pair (l, l^16) covers one row)
  float m = -3.0e38f, ssum = 0.f;
  for (int t = 0; t < 512; ++t) {
    const int j = c0 + (t << 1);
    const float sc = sr[j] * bf2f(mr[j]);
    if (sc > m) { ssum = ssum * __expf(m - sc) + 1.f; m = sc; }
    else        { ssum += __expf(sc - m); }
  }
  const float mo = __shfl_xor(m, 16, 32);
  const float so = __shfl_xor(ssum, 16, 32);
  const float mm = fmaxf(m, mo);
  ssum = ssum * __expf(m - mm) + so * __expf(mo - mm);
  m = mm;
  if (lane < 16) rowsum[wave][lane] = ssum;   // same-wave LDS: in-order

  F32x8 acc[4];
  #pragma unroll
  for (int dc = 0; dc < 4; ++dc)
    #pragma unroll
    for (int e = 0; e < 8; ++e) acc[dc].f[e] = 0.f;

  const unsigned short* vbase = vT + (size_t)(b * 16 + head) * 64 * 1024;

  // Per-lane LDS byte offsets for the async V-tile copies (flat addr[31:0]
  // is the LDS offset per ISA 10.2).
  const unsigned vtoff0 = (unsigned)(uintptr_t)&Vt[wave][lane][0];
  const unsigned vtoff1 = (unsigned)(uintptr_t)&Vt[wave][lane + 32][0];

  for (int jt = 0; jt < 32; ++jt) {
    const int jb = jt * 32;

    // WAR: make sure last iteration's fragment ds_loads completed before the
    // async engine overwrites Vt (DScnt and ASYNCcnt are independent).
    asm volatile("s_wait_dscnt 0x0" ::: "memory");

    // V^T tile: lane copies rows d = lane, lane+32 (64B each) into LDS via
    // async global->LDS b128 copies; INST_OFFSET applies to both addresses.
    {
      const unsigned short* gp0 = vbase + (size_t)lane * 1024 + jb;
      const unsigned short* gp1 = vbase + (size_t)(lane + 32) * 1024 + jb;
      #pragma unroll
      for (int q = 0; q < 4; ++q) {
        asm volatile("global_load_async_to_lds_b128 %0, %1, off offset:%2"
                     :: "v"(vtoff0), "v"(gp0), "i"(q * 16) : "memory");
        asm volatile("global_load_async_to_lds_b128 %0, %1, off offset:%2"
                     :: "v"(vtoff1), "v"(gp1), "i"(q * 16) : "memory");
      }
    }

    // P tile (unnormalized probs) -> LDS bf16 (overlaps with async copies)
    #pragma unroll
    for (int t = 0; t < 16; ++t) {
      const int jl = c0 + (t << 1);
      const float sc = sr[jb + jl] * bf2f(mr[jb + jl]);
      Pf[wave][r][jl] = f2bf(__expf(sc - m));
    }

    // B' = P^T (32j x 16i): N = i = lane&15, K = j
    BF16Frag bfr;
    #pragma unroll
    for (int hp = 0; hp < 8; ++hp)
      bfr.u[hp] = *(const unsigned*)&Pf[wave][r][frag_k(hp, c0)];

    // Wait for the async V-tile copies to land in LDS.
    asm volatile("s_wait_asynccnt 0x0" ::: "memory");

    // A' = V^T (16d x 32j) per d-chunk
    #pragma unroll
    for (int dc = 0; dc < 4; ++dc) {
      BF16Frag afr;
      const int dl = dc * 16 + r;
      #pragma unroll
      for (int hp = 0; hp < 8; ++hp)
        afr.u[hp] = *(const unsigned*)&Vt[wave][dl][frag_k(hp, c0)];
      acc[dc].v = __builtin_amdgcn_wmma_f32_16x16x32_bf16(
          false, afr.v, false, bfr.v, (short)0, acc[dc].v, false, false);
    }
  }

  // Epilogue: D layout (M=d= r8+8*hi, N=i=lane&15); normalize by row sum
  const float inv = 1.f / rowsum[wave][r];
  #pragma unroll
  for (int dc = 0; dc < 4; ++dc)
    #pragma unroll
    for (int r8 = 0; r8 < 8; ++r8) {
      const int d = dc * 16 + r8 + 8 * c0;
      const int i = i0 + r;
      attn_vec[(size_t)(i * 4 + b) * 1024 + head * 64 + d] = acc[dc].f[r8] * inv;
    }
}

// ---------------------------------------------------------------------------
// Residual + LayerNorm: out = LN(w + attn_out) * gamma + beta. One block/row.
// ---------------------------------------------------------------------------
__global__ __launch_bounds__(256)
void fishpp_ln(const float* __restrict__ w, const float* __restrict__ ao,
               const float* __restrict__ gamma, const float* __restrict__ beta,
               float* __restrict__ out)
{
  const int row = blockIdx.x;
  const float* wr = w  + (size_t)row * 1024;
  const float* ar = ao + (size_t)row * 1024;
  float x[4], s = 0.f, s2 = 0.f;
  #pragma unroll
  for (int t = 0; t < 4; ++t) {
    const int c = threadIdx.x + 256 * t;
    const float v = wr[c] + ar[c];
    x[t] = v; s += v; s2 += v * v;
  }
  #pragma unroll
  for (int o = 16; o > 0; o >>= 1) { s += __shfl_xor(s, o, 32); s2 += __shfl_xor(s2, o, 32); }
  __shared__ float ls[8], ls2[8];
  const int wave = threadIdx.x >> 5, lane = threadIdx.x & 31;
  if (lane == 0) { ls[wave] = s; ls2[wave] = s2; }
  __syncthreads();
  if (wave == 0) {
    float a = (lane < 8) ? ls[lane] : 0.f, b2 = (lane < 8) ? ls2[lane] : 0.f;
    #pragma unroll
    for (int o = 4; o > 0; o >>= 1) { a += __shfl_xor(a, o, 32); b2 += __shfl_xor(b2, o, 32); }
    if (lane == 0) { ls[0] = a; ls2[0] = b2; }
  }
  __syncthreads();
  const float mean = ls[0] * (1.f / 1024.f);
  const float var  = ls2[0] * (1.f / 1024.f) - mean * mean;
  const float rstd = rsqrtf(var + 1e-5f);
  #pragma unroll
  for (int t = 0; t < 4; ++t) {
    const int c = threadIdx.x + 256 * t;
    out[(size_t)row * 1024 + c] = (x[t] - mean) * rstd * gamma[c] + beta[c];
  }
}

// ---------------------------------------------------------------------------
// Host orchestration
// ---------------------------------------------------------------------------
extern "C" void kernel_launch(void* const* d_in, const int* in_sizes, int n_in,
                              void* d_out, int out_size, void* d_ws, size_t ws_size,
                              hipStream_t stream)
{
  (void)in_sizes; (void)n_in; (void)out_size;
  const float* w     = (const float*)d_in[0];
  const float* r     = (const float*)d_in[1];
  const float* rwb   = (const float*)d_in[2];
  const float* rrb   = (const float*)d_in[3];
  const float* masks = (const float*)d_in[4];
  const float* Wqkv  = (const float*)d_in[5];
  const float* Wr    = (const float*)d_in[6];
  const float* mproj = (const float*)d_in[7];
  const float* Wo    = (const float*)d_in[8];
  const float* gamma = (const float*)d_in[9];
  const float* beta  = (const float*)d_in[10];
  float* out = (float*)d_out;

  // Workspace layout (bytes); region 0 is reused: C1 -> BDraw -> attn_vec/C3
  const size_t OFF_R0  = 0;                       // max(C1 20.97MB, BDraw 32MB)
  const size_t OFF_QRW = 33554432;                // 2MB
  const size_t OFF_QRR = 35651584;                // 2MB
  const size_t OFF_K   = 37748736;                // 2MB
  const size_t OFF_RK  = 39845888;                // 0.5MB
  const size_t OFF_VT  = 40370176;                // 8MB bf16
  const size_t OFF_S   = 48758784;                // 32MB (C2 aliased at start)
  const size_t OFF_MW  = 82313216;                // 32MB bf16
  const size_t WS_NEED = 115867648;
  if (ws_size < WS_NEED) return;

  char* ws = (char*)d_ws;
  float* C1       = (float*)(ws + OFF_R0);
  float* BDraw    = (float*)(ws + OFF_R0);
  float* attn_vec = (float*)(ws + OFF_R0);
  float* C3       = (float*)(ws + OFF_R0 + 16777216);
  float* qrw      = (float*)(ws + OFF_QRW);
  float* qrr      = (float*)(ws + OFF_QRR);
  float* kbuf     = (float*)(ws + OFF_K);
  float* rk       = (float*)(ws + OFF_RK);
  unsigned short* vT = (unsigned short*)(ws + OFF_VT);
  float* C2       = (float*)(ws + OFF_S);
  float* S        = (float*)(ws + OFF_S);
  unsigned short* mwb = (unsigned short*)(ws + OFF_MW);

  // 1) w_heads = w @ W_qkv^T : [4096,1280]
  fishpp_gemm_bf16<<<dim3(10, 32, 1), 256, 0, stream>>>(
      w, 0, Wqkv, 0, 1, 1, C1, 0, 4096, 1280, 1024, 0, C1, 0);
  // 2) r_proj = r @ W_r^T : [1024,128]
  fishpp_gemm_bf16<<<dim3(1, 8, 1), 256, 0, stream>>>(
      r, 0, Wr, 0, 1, 1, C2, 0, 1024, 128, 1024, 0, C2, 0);
  // 3) scatter q/k/v (+biases, V transposed bf16)
  fishpp_scatter_qkv<<<dim3(20480), 256, 0, stream>>>(C1, rwb, rrb, qrw, qrr, kbuf, vT);
  // 4) scatter r_head_k
  fishpp_scatter_r<<<dim3(512), 256, 0, stream>>>(C2, rk);
  // 5) mw = SCALE * masks @ mask_proj  (bf16, [16][1024][1024])
  fishpp_mw<<<dim3(65536), 256, 0, stream>>>(masks, mproj, mwb);
  // 6) BDraw[b,g] = qrr[b,g] @ rk[g]^T : batch 8, K=64
  fishpp_gemm_bf16<<<dim3(8, 8, 8), 256, 0, stream>>>(
      qrr, 65536, rk, 65536, 1, 2, BDraw, 1048576, 1024, 1024, 64, 0, BDraw, 0);
  // 7) S[b,g] = qrw[b,g] @ k[b,g]^T + rel_shift(BDraw)  (mode 1)
  fishpp_gemm_bf16<<<dim3(8, 8, 8), 256, 0, stream>>>(
      qrw, 65536, kbuf, 65536, 1, 8, S, 1048576, 1024, 1024, 64, 1, BDraw, 1048576);
  // 8) fused mask-mix + softmax + PV -> attn_vec [4096,1024] fp32
  fishpp_attn<<<dim3(64, 2, 4), 256, 0, stream>>>(S, mwb, vT, attn_vec);
  // 9) attn_out = attn_vec @ W_o^T : [4096,1024]
  fishpp_gemm_bf16<<<dim3(8, 32, 1), 256, 0, stream>>>(
      attn_vec, 0, Wo, 0, 1, 1, C3, 0, 4096, 1024, 1024, 0, C3, 0);
  // 10) out = LayerNorm(w + attn_out)
  fishpp_ln<<<dim3(4096), 256, 0, stream>>>(w, C3, gamma, beta, out);
}
